// RNNT_66855460929943
// MI455X (gfx1250) — matrix operand order repaired and checked by reading, not compiled, not run on hardware
//
#include <hip/hip_runtime.h>
#include <hip/hip_bf16.h>

// ---------------------------------------------------------------------------
// RNN-T forward for MI455X (gfx1250).
//  * All GEMMs use v_wmma_f32_16x16x32_bf16 (bf16 operands, fp32 accumulate).
//  * LSTM recurrence: ONE fused kernel per step — WMMA h·Whh^T with the four
//    gate column-groups as the wave's four accumulators, LSTM nonlinearity in
//    the epilogue (no gate round-trip through HBM, half the launch chain).
//  * Final 334 MB logits tensor: pure float4 streaming stores (HBM-bound,
//    ~14 us at 23.3 TB/s — the irreducible cost of this problem).
// Workspace requirement: ~66 MB.
// ---------------------------------------------------------------------------

#define DI __device__ __forceinline__

typedef __attribute__((ext_vector_type(16))) __bf16 v16bf;
typedef __attribute__((ext_vector_type(8)))  float  v8f;

constexpr int B = 8, T = 200, F = 240, U = 50, U1 = 51;
constexpr int HE = 1024, HP = 640, E = 640, VOUT = 1024;
constexpr int FPAD = 256;   // F=240 padded to a multiple of 32 for K-stepping

DI __bf16 f2bf(float x) {
    // round-to-nearest-even f32 -> bf16
    unsigned u = __builtin_bit_cast(unsigned, x);
    unsigned r = u + 0x7FFFu + ((u >> 16) & 1u);
    unsigned short h = (unsigned short)(r >> 16);
    return __builtin_bit_cast(__bf16, h);
}
DI float sigm(float x) { return 1.0f / (1.0f + __expf(-x)); }

union Frag { v16bf v; uint4 q[2]; };

// Load one A fragment (16x32 bf16, lane half h holds contiguous K 16h..16h+15) from a K-major row.
DI v16bf load_afrag(const __bf16* arow, int k0, int ah) {
    Frag a;
    a.q[0] = *(const uint4*)(arow + k0 + ah);        // K = ah .. ah+7
    a.q[1] = *(const uint4*)(arow + k0 + ah + 8);    // K = ah+8 .. ah+15
    return a.v;
}
// Load one B fragment (32x16 bf16): lane holds 16 contiguous K of its column.
DI v16bf load_bfrag(const __bf16* wrow, int k0) {
    Frag b;
    b.q[0] = *(const uint4*)(wrow + k0);
    b.q[1] = *(const uint4*)(wrow + k0 + 8);
    return b.v;
}

// ---------------------------------------------------------------------------
// Generic GEMM: C[M,N] = A[M,K](bf16) * W[N,K]^T(bf16) (+bias[N]), fp32 out.
// One wave computes a 16x64 C tile (4 WMMA accumulators), K stepped by 32.
// Requires K % 32 == 0, N % 64 == 0.
// ---------------------------------------------------------------------------
__global__ void gemm_bf16_wmma(const __bf16* __restrict__ A,
                               const __bf16* __restrict__ W,
                               const float*  __restrict__ bias,
                               float*        __restrict__ C,
                               int M, int N, int K)
{
    const int wave = threadIdx.x >> 5;
    const int lane = threadIdx.x & 31;
    const int n0   = (blockIdx.x << 8) + (wave << 6);
    if (n0 >= N) return;                       // wave-uniform: EXEC stays full
    const int mt   = blockIdx.y << 4;
    int mrow = mt + (lane & 15);
    if (mrow >= M) mrow = M - 1;               // clamp ragged M (loads only)
    const int ah   = (lane < 16) ? 0 : 16;     // A half-K base per lane group
    const int bh   = (lane < 16) ? 0 : 16;     // B half-K base per lane group
    const int ncol = lane & 15;

    const __bf16* arow = A + (size_t)mrow * K;
    const __bf16* w0 = W + (size_t)(n0 +  0 + ncol) * K + bh;
    const __bf16* w1 = W + (size_t)(n0 + 16 + ncol) * K + bh;
    const __bf16* w2 = W + (size_t)(n0 + 32 + ncol) * K + bh;
    const __bf16* w3 = W + (size_t)(n0 + 48 + ncol) * K + bh;

    v8f acc0 = {}, acc1 = {}, acc2 = {}, acc3 = {};

    for (int k0 = 0; k0 < K; k0 += 32) {
        __builtin_prefetch(w0 + k0 + 128, 0, 1);     // global_prefetch_b8
        __builtin_prefetch(arow + k0 + 128, 0, 3);
        v16bf a  = load_afrag(arow, k0, ah);
        v16bf b0 = load_bfrag(w0, k0);
        v16bf b1 = load_bfrag(w1, k0);
        v16bf b2 = load_bfrag(w2, k0);
        v16bf b3 = load_bfrag(w3, k0);
        acc0 = __builtin_amdgcn_wmma_f32_16x16x32_bf16(false, a, false, b0, (short)0, acc0, false, false);
        acc1 = __builtin_amdgcn_wmma_f32_16x16x32_bf16(false, a, false, b1, (short)0, acc1, false, false);
        acc2 = __builtin_amdgcn_wmma_f32_16x16x32_bf16(false, a, false, b2, (short)0, acc2, false, false);
        acc3 = __builtin_amdgcn_wmma_f32_16x16x32_bf16(false, a, false, b3, (short)0, acc3, false, false);
    }

    // C/D 16x16: VGPR d -> row mt+d (lanes 0-15) or mt+8+d (lanes 16-31)
    const int rb = mt + ((lane < 16) ? 0 : 8);
    float bi0 = 0.f, bi1 = 0.f, bi2 = 0.f, bi3 = 0.f;
    if (bias) {
        bi0 = bias[n0 + ncol];      bi1 = bias[n0 + 16 + ncol];
        bi2 = bias[n0 + 32 + ncol]; bi3 = bias[n0 + 48 + ncol];
    }
#pragma unroll
    for (int d = 0; d < 8; ++d) {
        const int r = rb + d;
        if (r < M) {
            const size_t ro = (size_t)r * N + n0 + ncol;
            C[ro]      = acc0[d] + bi0;  C[ro + 16] = acc1[d] + bi1;
            C[ro + 32] = acc2[d] + bi2;  C[ro + 48] = acc3[d] + bi3;
        }
    }
}

// ---------------------------------------------------------------------------
// Fused LSTM step: gates = Hprev·Whh^T + Xg ; c' = sig(f)c + sig(i)tanh(g) ;
// h' = sig(o)tanh(c'). Each wave owns ONE 16-column group, and its four WMMA
// accumulators are that group at the four gate offsets {0,H,2H,3H} — so the
// full nonlinearity is applied in-register in the epilogue.
// Grid: H/16 waves, 4 waves/block. Requires H % 64 == 0 (1024, 640 are).
// ---------------------------------------------------------------------------
__global__ void lstm_step_fused(const __bf16* __restrict__ Hprev, // [B,H] bf16
                                const __bf16* __restrict__ Whh,   // [4H,H] bf16
                                const float*  __restrict__ Xg,    // [B,4H] pre-gates (incl bias)
                                float*        __restrict__ c,     // [B,H] fp32 state
                                __bf16*       __restrict__ hout,  // [B,H] bf16 history slice
                                int H)
{
    const int wave = threadIdx.x >> 5;
    const int lane = threadIdx.x & 31;
    const int n0   = ((blockIdx.x << 2) + wave) << 4;   // 16 cols per wave
    if (n0 >= H) return;
    int mrow = lane & 15;
    if (mrow >= B) mrow = B - 1;
    const int ah   = (lane < 16) ? 0 : 16;
    const int bh   = (lane < 16) ? 0 : 16;
    const int ncol = lane & 15;

    const __bf16* arow = Hprev + (size_t)mrow * H;
    const __bf16* wi = Whh + (size_t)(0 * H + n0 + ncol) * H + bh;
    const __bf16* wf = Whh + (size_t)(1 * H + n0 + ncol) * H + bh;
    const __bf16* wg = Whh + (size_t)(2 * H + n0 + ncol) * H + bh;
    const __bf16* wo = Whh + (size_t)(3 * H + n0 + ncol) * H + bh;

    v8f ai = {}, af = {}, ag = {}, ao = {};
    for (int k0 = 0; k0 < H; k0 += 32) {
        v16bf a  = load_afrag(arow, k0, ah);
        v16bf bi = load_bfrag(wi, k0);
        v16bf bf = load_bfrag(wf, k0);
        v16bf bg = load_bfrag(wg, k0);
        v16bf bo = load_bfrag(wo, k0);
        ai = __builtin_amdgcn_wmma_f32_16x16x32_bf16(false, a, false, bi, (short)0, ai, false, false);
        af = __builtin_amdgcn_wmma_f32_16x16x32_bf16(false, a, false, bf, (short)0, af, false, false);
        ag = __builtin_amdgcn_wmma_f32_16x16x32_bf16(false, a, false, bg, (short)0, ag, false, false);
        ao = __builtin_amdgcn_wmma_f32_16x16x32_bf16(false, a, false, bo, (short)0, ao, false, false);
    }

    const int rb = (lane < 16) ? 0 : 8;   // rows 8..15 are beyond B=8: no-op
#pragma unroll
    for (int d = 0; d < 8; ++d) {
        const int r = rb + d;
        if (r < B) {
            const int v = n0 + ncol;
            const float* xr = Xg + (size_t)r * 4 * H;
            const float ig = sigm(ai[d] + xr[v]);
            const float fg = sigm(af[d] + xr[H + v]);
            const float gg = tanhf(ag[d] + xr[2 * H + v]);
            const float og = sigm(ao[d] + xr[3 * H + v]);
            const size_t ci = (size_t)r * H + v;
            const float cn = fg * c[ci] + ig * gg;
            c[ci] = cn;
            hout[ci] = f2bf(og * tanhf(cn));
        }
    }
}

// --------------------------- elementwise kernels ---------------------------

__global__ void k_zero_u32(unsigned* __restrict__ p, int n) {
    int i = blockIdx.x * blockDim.x + threadIdx.x;
    if (i < n) p[i] = 0u;
}

// audio [B,1,F,T] -> X [T*B, FPAD] bf16 (time-major rows, zero-padded K)
__global__ void k_audio_pack(const float* __restrict__ audio, __bf16* __restrict__ X)
{
    long i = (long)blockIdx.x * blockDim.x + threadIdx.x;
    const long total = (long)T * B * FPAD;
    if (i >= total) return;
    int f = (int)(i % FPAD);
    long r = i / FPAD;
    int b = (int)(r % B);
    int t = (int)(r / B);
    float v = 0.f;
    if (f < F) v = audio[((long)b * F + f) * T + t];
    X[i] = f2bf(v);
}

// labels -> Y [U1*B, E] bf16, row u*B+b; u==0 is the zero start token
__global__ void k_embed_pack(const float* __restrict__ emb,
                             const int* __restrict__ labels,
                             __bf16* __restrict__ Y)
{
    long i = (long)blockIdx.x * blockDim.x + threadIdx.x;
    const long total = (long)U1 * B * E;
    if (i >= total) return;
    int e = (int)(i % E);
    long r = i / E;
    int b = (int)(r % B);
    int u = (int)(r / B);
    float v = 0.f;
    if (u > 0) {
        int lab = labels[b * U + (u - 1)];
        v = emb[(long)lab * E + e];
    }
    Y[i] = f2bf(v);
}

// generic f32 -> bf16 with K padding (kin -> kout)
__global__ void k_cvt_pad(const float* __restrict__ in, __bf16* __restrict__ out,
                          int rows, int kin, int kout)
{
    long i = (long)blockIdx.x * blockDim.x + threadIdx.x;
    const long total = (long)rows * kout;
    if (i >= total) return;
    int k = (int)(i % kout);
    long r = i / kout;
    out[i] = f2bf(k < kin ? in[r * kin + k] : 0.f);
}

// fc_W [VOUT, HE+HP] -> We [VOUT,HE] bf16, Wp [VOUT,HP] bf16
__global__ void k_fc_split(const float* __restrict__ fcW,
                           __bf16* __restrict__ We, __bf16* __restrict__ Wp)
{
    long i = (long)blockIdx.x * blockDim.x + threadIdx.x;
    const long total = (long)VOUT * (HE + HP);
    if (i >= total) return;
    int h = (int)(i % (HE + HP));
    int v = (int)(i / (HE + HP));
    float val = fcW[i];
    if (h < HE) We[(long)v * HE + h] = f2bf(val);
    else        Wp[(long)v * HP + (h - HE)] = f2bf(val);
}

// logits[b,t,u,:] = pf[t*B+b,:] + pg[u*B+b,:] + fc_b  (float4 streaming)
__global__ void k_joint(const float4* __restrict__ pf, const float4* __restrict__ pg,
                        const float4* __restrict__ fcb, float4* __restrict__ out,
                        long total4)
{
    long i = (long)blockIdx.x * blockDim.x + threadIdx.x;
    if (i >= total4) return;
    int v4 = (int)(i & (VOUT / 4 - 1));   // VOUT/4 == 256
    long r = i >> 8;                      // (b*T + t)*U1 + u
    int u = (int)(r % U1);
    long bt = r / U1;
    int t = (int)(bt % T);
    int b = (int)(bt / T);
    float4 a = pf[((long)(t * B + b) << 8) + v4];
    float4 c = pg[((long)(u * B + b) << 8) + v4];
    float4 e = fcb[v4];
    float4 o;
    o.x = a.x + c.x + e.x;  o.y = a.y + c.y + e.y;
    o.z = a.z + c.z + e.z;  o.w = a.w + c.w + e.w;
    out[i] = o;
}

// ------------------------------- launcher ----------------------------------

extern "C" void kernel_launch(void* const* d_in, const int* in_sizes, int n_in,
                              void* d_out, int out_size, void* d_ws, size_t ws_size,
                              hipStream_t stream)
{
    (void)in_sizes; (void)n_in; (void)out_size; (void)ws_size;

    const float* audio   = (const float*)d_in[0];
    const int*   labels  = (const int*)  d_in[1];
    // d_in[2]/d_in[3] (audio_lens/label_lens) are unused by the reference math
    const float* enc_Wih = (const float*)d_in[4];
    const float* enc_Whh = (const float*)d_in[5];
    const float* enc_b   = (const float*)d_in[6];
    const float* emb     = (const float*)d_in[7];
    const float* dec_Wih = (const float*)d_in[8];
    const float* dec_Whh = (const float*)d_in[9];
    const float* dec_b   = (const float*)d_in[10];
    const float* fc_W    = (const float*)d_in[11];
    const float* fc_b    = (const float*)d_in[12];
    float* out = (float*)d_out;

    // ---- workspace carve-up (deterministic arithmetic each call) ----
    char* p = (char*)d_ws;
    auto take = [&](size_t bytes) -> char* {
        char* r = p; p += (bytes + 255) & ~(size_t)255; return r;
    };
    float*  XgEnc   = (float*) take((size_t)T * B * 4 * HE * 4);   // 26.2 MB
    float*  XgDec   = (float*) take((size_t)U1 * B * 4 * HP * 4);  //  4.2 MB
    float*  cEnc    = (float*) take((size_t)B * HE * 4);
    float*  cDec    = (float*) take((size_t)B * HP * 4);
    __bf16* fbf     = (__bf16*)take((size_t)T * B * HE * 2);       //  3.3 MB
    __bf16* gbf     = (__bf16*)take((size_t)U1 * B * HP * 2);
    __bf16* Xbf     = (__bf16*)take((size_t)T * B * FPAD * 2);
    __bf16* Ybf     = (__bf16*)take((size_t)U1 * B * E * 2);
    __bf16* WihE    = (__bf16*)take((size_t)4 * HE * FPAD * 2);
    __bf16* WhhE    = (__bf16*)take((size_t)4 * HE * HE * 2);      //  8.4 MB
    __bf16* WihD    = (__bf16*)take((size_t)4 * HP * E * 2);
    __bf16* WhhD    = (__bf16*)take((size_t)4 * HP * HP * 2);
    __bf16* Webf    = (__bf16*)take((size_t)VOUT * HE * 2);
    __bf16* Wpbf    = (__bf16*)take((size_t)VOUT * HP * 2);
    float*  pf      = (float*) take((size_t)T * B * VOUT * 4);     //  6.6 MB
    float*  pg      = (float*) take((size_t)U1 * B * VOUT * 4);
    __bf16* zbf     = (__bf16*)take((size_t)16 * 1024 * 2);        // zero h0

    auto g1 = [](long n) { return dim3((unsigned)((n + 255) / 256)); };
    auto gemm_grid = [](int M, int N) { return dim3((unsigned)((N + 255) / 256),
                                                    (unsigned)((M + 15) / 16)); };

    // ---- per-call state init (deterministic; required post-poison) ----
    k_zero_u32<<<g1(B * HE), 256, 0, stream>>>((unsigned*)cEnc, B * HE);
    k_zero_u32<<<g1(B * HP), 256, 0, stream>>>((unsigned*)cDec, B * HP);
    k_zero_u32<<<g1(16 * 1024 / 2), 256, 0, stream>>>((unsigned*)zbf, 16 * 1024 / 2);

    // ---- pack / convert to bf16 ----
    k_audio_pack<<<g1((long)T * B * FPAD), 256, 0, stream>>>(audio, Xbf);
    k_embed_pack<<<g1((long)U1 * B * E), 256, 0, stream>>>(emb, labels, Ybf);
    k_cvt_pad<<<g1((long)4 * HE * FPAD), 256, 0, stream>>>(enc_Wih, WihE, 4 * HE, F, FPAD);
    k_cvt_pad<<<g1((long)4 * HE * HE),   256, 0, stream>>>(enc_Whh, WhhE, 4 * HE, HE, HE);
    k_cvt_pad<<<g1((long)4 * HP * E),    256, 0, stream>>>(dec_Wih, WihD, 4 * HP, E, E);
    k_cvt_pad<<<g1((long)4 * HP * HP),   256, 0, stream>>>(dec_Whh, WhhD, 4 * HP, HP, HP);
    k_fc_split<<<g1((long)VOUT * (HE + HP)), 256, 0, stream>>>(fc_W, Webf, Wpbf);

    // ---- hoisted input projections: Xg = X*Wih^T + b for every timestep ----
    gemm_bf16_wmma<<<gemm_grid(T * B, 4 * HE), 128, 0, stream>>>(
        Xbf, WihE, enc_b, XgEnc, T * B, 4 * HE, FPAD);
    gemm_bf16_wmma<<<gemm_grid(U1 * B, 4 * HP), 128, 0, stream>>>(
        Ybf, WihD, dec_b, XgDec, U1 * B, 4 * HP, E);

    // ---- encoder recurrence: 200 chained fused steps (WMMA + cell) ----
    for (int t = 0; t < T; ++t) {
        const __bf16* hprev = (t == 0) ? zbf : (fbf + (size_t)(t - 1) * B * HE);
        lstm_step_fused<<<dim3(HE / 64), 128, 0, stream>>>(
            hprev, WhhE, XgEnc + (size_t)t * B * 4 * HE,
            cEnc, fbf + (size_t)t * B * HE, HE);
    }

    // ---- decoder recurrence: 51 chained fused steps ----
    for (int u = 0; u < U1; ++u) {
        const __bf16* hprev = (u == 0) ? zbf : (gbf + (size_t)(u - 1) * B * HP);
        lstm_step_fused<<<dim3(HP / 64), 128, 0, stream>>>(
            hprev, WhhD, XgDec + (size_t)u * B * 4 * HP,
            cDec, gbf + (size_t)u * B * HP, HP);
    }

    // ---- joint projections ----
    gemm_bf16_wmma<<<gemm_grid(T * B, VOUT), 128, 0, stream>>>(
        fbf, Webf, nullptr, pf, T * B, VOUT, HE);
    gemm_bf16_wmma<<<gemm_grid(U1 * B, VOUT), 128, 0, stream>>>(
        gbf, Wpbf, nullptr, pg, U1 * B, VOUT, HP);

    // ---- broadcast add: 334 MB streaming float4 stores (BW-bound) ----
    const long total4 = (long)B * T * U1 * VOUT / 4;
    k_joint<<<g1(total4), 256, 0, stream>>>(
        (const float4*)pf, (const float4*)pg, (const float4*)fc_b,
        (float4*)out, total4);
}
